// MoonVitEncoderLayer_18545668784314
// MI455X (gfx1250) — matrix operand order repaired and checked by reading, not compiled, not run on hardware
//
#include <hip/hip_runtime.h>

// ---------------------------------------------------------------------------
// MoonVit encoder layer for gfx1250 (CDNA5, wave32, WMMA bf16 16x16x32).
// GEMMs use async global->LDS staging (ASYNCcnt, double buffered) + WMMA.
// ---------------------------------------------------------------------------

#define TOTAL    3072
#define HIDDEN   1024
#define NUM_HEADS 16
#define HEAD_DIM  64
#define MLP_DIM  4096
#define SEG      1024
#define NSEG     3

typedef __attribute__((ext_vector_type(16))) __bf16 v16bf;
typedef __attribute__((ext_vector_type(8)))  __bf16 v8bf;
typedef __attribute__((ext_vector_type(8)))  float  v8f;
typedef __attribute__((ext_vector_type(4)))  int    v4i;

typedef __attribute__((address_space(1))) v4i gv4i;   // global
typedef __attribute__((address_space(3))) v4i lv4i;   // LDS

// A-matrix 16x32 bf16 fragment (per ISA layout):
//   lanes 0-15  hold row M=lane,      K = {0..7} U {16..23}
//   lanes 16-31 hold row M=lane-16,   K = {8..15} U {24..31}
// caller passes p = &A[row][k0 + half*8]; we grab 8 elems at +0 and 8 at +16.
__device__ __forceinline__ v16bf load_frag_a(const __bf16* p) {
  v8bf lo = *(const v8bf*)(p);
  v8bf hi = *(const v8bf*)(p + 16);
  return __builtin_shufflevector(lo, hi, 0,1,2,3,4,5,6,7,8,9,10,11,12,13,14,15);
}

// B-matrix 32x16 bf16 fragment (per ISA layout):
//   lanes 0-15  hold col N=lane,      K = {0..7} U {16..23}
//   lanes 16-31 hold col N=lane-16,   K = {8..15} U {24..31}
// caller passes p = &Bt[col][k0 + half*8]; we grab 8 elems at +0 and 8 at +16.
__device__ __forceinline__ v16bf load_frag_b(const __bf16* p) {
  v8bf lo = *(const v8bf*)(p);
  v8bf hi = *(const v8bf*)(p + 16);
  return __builtin_shufflevector(lo, hi, 0,1,2,3,4,5,6,7,8,9,10,11,12,13,14,15);
}

__device__ __forceinline__ v8f wmma_bf16(v16bf a, v16bf b, v8f c) {
  return __builtin_amdgcn_wmma_f32_16x16x32_bf16(false, a, false, b, (short)0, c,
                                                 false, false);
}

__device__ __forceinline__ float gelu_exact(float x) {
  return 0.5f * x * (1.0f + erff(x * 0.70710678118654752440f));
}

// --- CDNA5 async global->LDS copy (GLOBAL_LOAD_ASYNC_TO_LDS_B128, ASYNCcnt) --
__device__ __forceinline__ void async_copy_b128(const void* gsrc, void* lds_dst) {
#if __has_builtin(__builtin_amdgcn_global_load_async_to_lds_b128)
  __builtin_amdgcn_global_load_async_to_lds_b128(
      (gv4i*)gsrc, (lv4i*)lds_dst, 0, 0);
#else
  unsigned laddr = (unsigned)(unsigned long long)lds_dst;
  asm volatile("global_load_async_to_lds_b128 %0, %1, off"
               :: "v"(laddr), "v"(gsrc) : "memory");
#endif
}

template <int N>
__device__ __forceinline__ void wait_asynccnt() {
#if __has_builtin(__builtin_amdgcn_s_wait_asynccnt)
  __builtin_amdgcn_s_wait_asynccnt(N);
#else
  asm volatile("s_wait_asynccnt %0" :: "i"(N) : "memory");
#endif
}

// ---------------------------------------------------------------------------
// Transpose + fp32->bf16 convert:  in[K x N] fp32 (row-major) -> out[N x K] bf16
// block = (32,8), 32x32 tiles via LDS. K,N multiples of 32.
// ---------------------------------------------------------------------------
__global__ __launch_bounds__(256)
void transpose_cvt_kernel(const float* __restrict__ in, __bf16* __restrict__ out,
                          int K, int N) {
  __shared__ float tile[32][33];
  int ntiles = N / 32;
  int bx = blockIdx.x % ntiles;          // column tile of input
  int by = blockIdx.x / ntiles;          // row tile of input
  int n0 = bx * 32, k0 = by * 32;
  int tx = threadIdx.x, ty = threadIdx.y;
#pragma unroll
  for (int r = 0; r < 4; ++r)
    tile[ty + r * 8][tx] = in[(size_t)(k0 + ty + r * 8) * N + n0 + tx];
  __syncthreads();
#pragma unroll
  for (int r = 0; r < 4; ++r)
    out[(size_t)(n0 + ty + r * 8) * K + k0 + tx] = (__bf16)tile[tx][ty + r * 8];
}

// ---------------------------------------------------------------------------
// LayerNorm over rows of 1024, fp32 in -> bf16 out. 256 threads per row.
// ---------------------------------------------------------------------------
__global__ __launch_bounds__(256)
void layernorm_kernel(const float* __restrict__ x, const float* __restrict__ w,
                      const float* __restrict__ b, __bf16* __restrict__ out) {
  __shared__ float red[256];
  int row = blockIdx.x, tid = threadIdx.x;
  const float* xr = x + (size_t)row * HIDDEN;
  float vals[4];
  float s = 0.f;
#pragma unroll
  for (int i = 0; i < 4; ++i) { vals[i] = xr[tid + i * 256]; s += vals[i]; }
  red[tid] = s; __syncthreads();
  for (int st = 128; st > 0; st >>= 1) {
    if (tid < st) red[tid] += red[tid + st];
    __syncthreads();
  }
  float mean = red[0] * (1.0f / HIDDEN);
  __syncthreads();
  float vs = 0.f;
#pragma unroll
  for (int i = 0; i < 4; ++i) { float d = vals[i] - mean; vs += d * d; }
  red[tid] = vs; __syncthreads();
  for (int st = 128; st > 0; st >>= 1) {
    if (tid < st) red[tid] += red[tid + st];
    __syncthreads();
  }
  float rstd = rsqrtf(red[0] * (1.0f / HIDDEN) + 1e-5f);
#pragma unroll
  for (int i = 0; i < 4; ++i) {
    int c = tid + i * 256;
    out[(size_t)row * HIDDEN + c] = (__bf16)((vals[i] - mean) * rstd * w[c] + b[c]);
  }
}

// ---------------------------------------------------------------------------
// NT GEMM: C[M x N] = A[M x K] (bf16, row-major) * Bt[N x K]^T (bf16, row-major)
// Workgroup (256 thr = 8 waves) computes a 128x128 tile. A(128x32) and B(128x32)
// K-slabs are staged in LDS with GLOBAL_LOAD_ASYNC_TO_LDS_B128, double-buffered
// via s_wait_asynccnt. Each wave computes a 32x64 sub-tile (2x4 WMMA accums).
// Fused epilogue: optional bias[N], GELU, fp32 residual, bf16 or f32 store.
// ---------------------------------------------------------------------------
__global__ __launch_bounds__(256)
void gemm_bf16_kernel(const __bf16* __restrict__ A, const __bf16* __restrict__ Bt,
                      int M, int N, int K,
                      const float* __restrict__ bias,
                      const float* __restrict__ residual,
                      int act,
                      __bf16* __restrict__ outb, float* __restrict__ outf) {
  __shared__ __bf16 Asb[2][128 * 32];   // 8 KB per buffer
  __shared__ __bf16 Bsb[2][128 * 32];   // 8 KB per buffer

  int tid = threadIdx.x;
  int wid = tid >> 5;
  int lane = tid & 31;
  int r15 = lane & 15, half = lane >> 4;

  int tilesN = N / 128;
  int bm = blockIdx.x / tilesN, bn = blockIdx.x % tilesN;
  int m0 = bm * 128, n0 = bn * 128;
  int wm = wid >> 1, wn = wid & 1;       // 4 x 2 wave grid

  // Per K-slab, each wave issues 2 async b128 for A and 2 for B.
  // slab = 128 rows x 32 cols bf16 = 8192 B = 16 wave-wide b128 transfers.
  auto prefetch = [&](int buf, int k0) {
#pragma unroll
    for (int c = 0; c < 2; ++c) {
      int u = (wid * 2 + c) * 32 + lane;     // 0..511 16-byte units
      int o = u * 16;                        // byte offset within slab
      int row = o >> 6;                      // 64 B per row (32 bf16)
      int col = o & 63;
      async_copy_b128((const char*)A + ((size_t)(m0 + row) * K + k0) * 2 + col,
                      (char*)&Asb[buf][0] + o);
      async_copy_b128((const char*)Bt + ((size_t)(n0 + row) * K + k0) * 2 + col,
                      (char*)&Bsb[buf][0] + o);
    }
  };

  const v8f vzero = {0.f, 0.f, 0.f, 0.f, 0.f, 0.f, 0.f, 0.f};
  v8f acc[2][4];
#pragma unroll
  for (int i = 0; i < 2; ++i)
#pragma unroll
    for (int j = 0; j < 4; ++j) acc[i][j] = vzero;

  int nsteps = K / 32;
  prefetch(0, 0);
  for (int ks = 0; ks < nsteps; ++ks) {
    if (ks + 1 < nsteps) {
      prefetch((ks + 1) & 1, (ks + 1) * 32);
      wait_asynccnt<4>();                 // previous slab's 4 ops complete
    } else {
      wait_asynccnt<0>();
    }
    __syncthreads();

    const __bf16* As = &Asb[ks & 1][0];
    const __bf16* Bs = &Bsb[ks & 1][0];
    v16bf af[2], bfr[4];
#pragma unroll
    for (int i = 0; i < 2; ++i) {
      int row = wm * 32 + i * 16 + r15;
      af[i] = load_frag_a(As + row * 32 + half * 8);
    }
#pragma unroll
    for (int j = 0; j < 4; ++j) {
      int row = wn * 64 + j * 16 + r15;
      bfr[j] = load_frag_b(Bs + row * 32 + half * 8);
    }
#pragma unroll
    for (int i = 0; i < 2; ++i)
#pragma unroll
      for (int j = 0; j < 4; ++j) acc[i][j] = wmma_bf16(af[i], bfr[j], acc[i][j]);

    __syncthreads();                      // all reads done before overwrite
  }

  int wm0 = m0 + wm * 32, wn0 = n0 + wn * 64;
#pragma unroll
  for (int i = 0; i < 2; ++i) {
#pragma unroll
    for (int j = 0; j < 4; ++j) {
      int col = wn0 + j * 16 + r15;
      float bv = bias ? bias[col] : 0.f;
#pragma unroll
      for (int r = 0; r < 8; ++r) {
        int row = wm0 + i * 16 + r + half * 8;
        float v = acc[i][j][r] + bv;
        if (act == 1) v = gelu_exact(v);
        if (residual) v += residual[(size_t)row * N + col];
        if (outb) outb[(size_t)row * N + col] = (__bf16)v;
        else      outf[(size_t)row * N + col] = v;
      }
    }
  }
}

// ---------------------------------------------------------------------------
// RoPE + QKV split. qkv is bf16 [token][3*HIDDEN] with col = qkv*1024+head*64+d.
// Writes qb/kb bf16 [token][1024] (rotated) and vT bf16 [head*64+d][token].
// ---------------------------------------------------------------------------
__global__ __launch_bounds__(256)
void rope_split_kernel(const __bf16* __restrict__ qkv,
                       const float* __restrict__ rcos, const float* __restrict__ rsin,
                       __bf16* __restrict__ qb, __bf16* __restrict__ kb,
                       __bf16* __restrict__ vT) {
  int idx = blockIdx.x * blockDim.x + threadIdx.x;
  if (idx >= TOTAL * NUM_HEADS * (HEAD_DIM / 2)) return;
  int i = idx & 31;
  int head = (idx >> 5) & 15;
  int t = idx >> 9;
  int col = head * HEAD_DIM + 2 * i;
  float c = rcos[t * 32 + i], s = rsin[t * 32 + i];
  size_t base = (size_t)t * (3 * HIDDEN);
  float a = (float)qkv[base + col];
  float b = (float)qkv[base + col + 1];
  qb[(size_t)t * HIDDEN + col]     = (__bf16)(a * c - b * s);
  qb[(size_t)t * HIDDEN + col + 1] = (__bf16)(a * s + b * c);
  a = (float)qkv[base + HIDDEN + col];
  b = (float)qkv[base + HIDDEN + col + 1];
  kb[(size_t)t * HIDDEN + col]     = (__bf16)(a * c - b * s);
  kb[(size_t)t * HIDDEN + col + 1] = (__bf16)(a * s + b * c);
  vT[(size_t)(col)     * TOTAL + t] = qkv[base + 2 * HIDDEN + col];
  vT[(size_t)(col + 1) * TOTAL + t] = qkv[base + 2 * HIDDEN + col + 1];
}

// ---------------------------------------------------------------------------
// Attention: one wave (32 threads) per (head, segment, 16-query tile).
// S = Q*K^T (WMMA) -> 16x1024 f32 LDS, per-row softmax (shfl_xor butterflies),
// P*V (WMMA, P read from LDS as bf16, V from vT). Mask implicit via segment.
// ---------------------------------------------------------------------------
__global__ __launch_bounds__(32)
void attention_kernel(const __bf16* __restrict__ qb, const __bf16* __restrict__ kb,
                      const __bf16* __restrict__ vT, __bf16* __restrict__ attn) {
  __shared__ float S[16 * SEG];           // 64 KB
  int blk = blockIdx.x;
  int qt   = blk & 63;
  int seg  = (blk >> 6) % NSEG;
  int head = blk / (64 * NSEG);
  int lane = threadIdx.x;
  int r15 = lane & 15, half = lane >> 4;
  int q0 = seg * SEG + qt * 16;
  int kbase = seg * SEG;
  const float scale = 0.125f;             // 1/sqrt(64)

  v16bf aq[2];
#pragma unroll
  for (int kk = 0; kk < 2; ++kk)
    aq[kk] = load_frag_a(qb + (size_t)(q0 + r15) * HIDDEN + head * HEAD_DIM +
                         kk * 32 + half * 8);

  const v8f vzero = {0.f, 0.f, 0.f, 0.f, 0.f, 0.f, 0.f, 0.f};

  for (int kt = 0; kt < SEG / 16; ++kt) {
    v8f acc = vzero;
#pragma unroll
    for (int kk = 0; kk < 2; ++kk) {
      v16bf bk = load_frag_b(kb + (size_t)(kbase + kt * 16 + r15) * HIDDEN +
                             head * HEAD_DIM + kk * 32 + half * 8);
      acc = wmma_bf16(aq[kk], bk, acc);
    }
#pragma unroll
    for (int r = 0; r < 8; ++r)
      S[(r + half * 8) * SEG + kt * 16 + r15] = acc[r] * scale;
  }
  __syncthreads();

  for (int m = 0; m < 16; ++m) {
    float* row = S + m * SEG;
    float mx = -1e30f;
    for (int j = lane; j < SEG; j += 32) mx = fmaxf(mx, row[j]);
#pragma unroll
    for (int off = 16; off > 0; off >>= 1) mx = fmaxf(mx, __shfl_xor(mx, off, 32));
    float sum = 0.f;
    for (int j = lane; j < SEG; j += 32) {
      float e = __expf(row[j] - mx);
      row[j] = e;
      sum += e;
    }
#pragma unroll
    for (int off = 16; off > 0; off >>= 1) sum += __shfl_xor(sum, off, 32);
    float inv = 1.0f / sum;
    for (int j = lane; j < SEG; j += 32) row[j] *= inv;
  }
  __syncthreads();

  v8f acc4[4];
#pragma unroll
  for (int j = 0; j < 4; ++j) acc4[j] = vzero;

  for (int ks = 0; ks < SEG / 32; ++ks) {
    v16bf ap;
#pragma unroll
    for (int j = 0; j < 16; ++j) {
      int kcol = ks * 32 + half * 8 + (j < 8 ? j : j + 8);
      ap[j] = (__bf16)S[r15 * SEG + kcol];
    }
#pragma unroll
    for (int nt = 0; nt < 4; ++nt) {
      const __bf16* vp = vT + (size_t)(head * HEAD_DIM + nt * 16 + r15) * TOTAL +
                         kbase + ks * 32 + half * 8;
      acc4[nt] = wmma_bf16(ap, load_frag_b(vp), acc4[nt]);
    }
  }

#pragma unroll
  for (int nt = 0; nt < 4; ++nt) {
    int col = head * HEAD_DIM + nt * 16 + r15;
#pragma unroll
    for (int r = 0; r < 8; ++r) {
      int row = q0 + r + half * 8;
      attn[(size_t)row * HIDDEN + col] = (__bf16)acc4[nt][r];
    }
  }
}

// ---------------------------------------------------------------------------
// Host-side orchestration
// ---------------------------------------------------------------------------
extern "C" void kernel_launch(void* const* d_in, const int* in_sizes, int n_in,
                              void* d_out, int out_size, void* d_ws, size_t ws_size,
                              hipStream_t stream) {
  const float* hidden  = (const float*)d_in[0];
  const float* rcos    = (const float*)d_in[1];
  const float* rsin    = (const float*)d_in[2];
  const float* n0w     = (const float*)d_in[3];
  const float* n0b     = (const float*)d_in[4];
  const float* n1w     = (const float*)d_in[5];
  const float* n1b     = (const float*)d_in[6];
  const float* wqkv    = (const float*)d_in[7];
  const float* wo      = (const float*)d_in[8];
  const float* fc0w    = (const float*)d_in[9];
  const float* fc0b    = (const float*)d_in[10];
  const float* fc1w    = (const float*)d_in[11];
  const float* fc1b    = (const float*)d_in[12];
  float* out = (float*)d_out;

  char* ws = (char*)d_ws;
  size_t off = 0;
  auto alloc = [&](size_t bytes) { char* p = ws + off; off = (off + bytes + 255) & ~(size_t)255; return p; };
  __bf16* wqkvT = (__bf16*)alloc((size_t)3 * HIDDEN * HIDDEN * 2);      // [3072][1024]
  __bf16* woT   = (__bf16*)alloc((size_t)HIDDEN * HIDDEN * 2);          // [1024][1024]
  __bf16* fc0T  = (__bf16*)alloc((size_t)MLP_DIM * HIDDEN * 2);         // [4096][1024]
  __bf16* fc1T  = (__bf16*)alloc((size_t)HIDDEN * MLP_DIM * 2);         // [1024][4096]
  __bf16* x0    = (__bf16*)alloc((size_t)TOTAL * HIDDEN * 2);
  __bf16* qkv   = (__bf16*)alloc((size_t)TOTAL * 3 * HIDDEN * 2);
  __bf16* qbuf  = (__bf16*)alloc((size_t)TOTAL * HIDDEN * 2);
  __bf16* kbuf  = (__bf16*)alloc((size_t)TOTAL * HIDDEN * 2);
  __bf16* vT    = (__bf16*)alloc((size_t)HIDDEN * TOTAL * 2);
  __bf16* attn  = (__bf16*)alloc((size_t)TOTAL * HIDDEN * 2);
  float*  hbuf  = (float*) alloc((size_t)TOTAL * HIDDEN * 4);
  __bf16* x2    = (__bf16*)alloc((size_t)TOTAL * HIDDEN * 2);
  __bf16* g     = (__bf16*)alloc((size_t)TOTAL * MLP_DIM * 2);

  dim3 tblk(32, 8);

  transpose_cvt_kernel<<<(3 * HIDDEN / 32) * (HIDDEN / 32), tblk, 0, stream>>>(
      wqkv, wqkvT, HIDDEN, 3 * HIDDEN);
  transpose_cvt_kernel<<<(HIDDEN / 32) * (HIDDEN / 32), tblk, 0, stream>>>(
      wo, woT, HIDDEN, HIDDEN);
  transpose_cvt_kernel<<<(MLP_DIM / 32) * (HIDDEN / 32), tblk, 0, stream>>>(
      fc0w, fc0T, HIDDEN, MLP_DIM);
  transpose_cvt_kernel<<<(HIDDEN / 32) * (MLP_DIM / 32), tblk, 0, stream>>>(
      fc1w, fc1T, MLP_DIM, HIDDEN);

  layernorm_kernel<<<TOTAL, 256, 0, stream>>>(hidden, n0w, n0b, x0);

  // QKV GEMM: [3072x1024] x [1024x3072] -> bf16 qkv
  gemm_bf16_kernel<<<(TOTAL / 128) * (3 * HIDDEN / 128), 256, 0, stream>>>(
      x0, wqkvT, TOTAL, 3 * HIDDEN, HIDDEN, nullptr, nullptr, 0, qkv, nullptr);

  {
    int n = TOTAL * NUM_HEADS * (HEAD_DIM / 2);
    rope_split_kernel<<<(n + 255) / 256, 256, 0, stream>>>(qkv, rcos, rsin, qbuf,
                                                           kbuf, vT);
  }

  attention_kernel<<<NUM_HEADS * NSEG * (SEG / 16), 32, 0, stream>>>(qbuf, kbuf, vT,
                                                                     attn);

  // h = residual + attn @ wo  (f32 out)
  gemm_bf16_kernel<<<(TOTAL / 128) * (HIDDEN / 128), 256, 0, stream>>>(
      attn, woT, TOTAL, HIDDEN, HIDDEN, nullptr, hidden, 0, nullptr, hbuf);

  layernorm_kernel<<<TOTAL, 256, 0, stream>>>(hbuf, n1w, n1b, x2);

  // FC0 + bias + GELU -> bf16
  gemm_bf16_kernel<<<(TOTAL / 128) * (MLP_DIM / 128), 256, 0, stream>>>(
      x2, fc0T, TOTAL, MLP_DIM, HIDDEN, fc0b, nullptr, 1, g, nullptr);

  // FC1 + bias + residual(h) -> f32 d_out
  gemm_bf16_kernel<<<(TOTAL / 128) * (HIDDEN / 128), 256, 0, stream>>>(
      g, fc1T, TOTAL, HIDDEN, MLP_DIM, fc1b, hbuf, 0, nullptr, out);
}
